// MixtureOfExperts_50483045597273
// MI455X (gfx1250) — matrix operand order repaired and checked by reading, not compiled, not run on hardware
//
#include <hip/hip_runtime.h>

// ---------------------------------------------------------------------------
// MoE on gfx1250: bf16 WMMA GEMMs (v_wmma_f32_16x16x32_bf16), wave32,
// double-buffered LDS (K-step 64), async global->LDS staging for the A tile.
// ---------------------------------------------------------------------------

typedef __attribute__((ext_vector_type(16))) __bf16 v16bf;
typedef __attribute__((ext_vector_type(8)))  float  v8f;
typedef __attribute__((ext_vector_type(4)))  unsigned int u32x4;

// Builtin-compatible vector-of-int pointee for async LDS copies:
typedef int i32x4 __attribute__((vector_size(16)));
typedef __attribute__((address_space(1))) i32x4 as1_i32x4;
typedef __attribute__((address_space(3))) i32x4 as3_i32x4;

union Frag16 {
    unsigned int u[8];
    v16bf v;
};

#if __has_builtin(__builtin_amdgcn_global_load_async_to_lds_b128)
#define MOE_HAVE_ASYNC_LDS 1
#else
#define MOE_HAVE_ASYNC_LDS 0
#endif

static __device__ __forceinline__ void moe_async_wait_all() {
#if MOE_HAVE_ASYNC_LDS
#if __has_builtin(__builtin_amdgcn_s_wait_asynccnt)
    __builtin_amdgcn_s_wait_asynccnt(0);
#else
    asm volatile("s_wait_asynccnt 0x0" ::: "memory");
#endif
#endif
}

static __device__ __forceinline__ unsigned short f2bf(float x) {
    unsigned int u = __builtin_bit_cast(unsigned int, x);
    unsigned int r = u + 0x7FFFu + ((u >> 16) & 1u);   // round-to-nearest-even
    return (unsigned short)(r >> 16);
}

// --- A fragment: 16x32 bf16 (rows striped per ISA table), LDS row stride 64 --
static __device__ __forceinline__ void load_a_frag(const unsigned short* __restrict__ smem,
                                                   int rowBase, int ksub, int lane, Frag16& f) {
    const int r    = rowBase + (lane & 15);
    const int koff = (lane >> 4) * 8 + ksub;
    const unsigned short* p = smem + r * 64 + koff;
#pragma unroll
    for (int i = 0; i < 8; ++i) {
        const int k = ((i >> 2) * 16) + ((i & 3) * 2);
        f.u[i] = *(const unsigned int*)(p + k);
    }
}

// --- B fragment: 32x16 bf16 from LDS stored transposed [N][K], stride 64 -----
static __device__ __forceinline__ void load_b_frag(const unsigned short* __restrict__ smemT,
                                                   int colBase, int ksub, int lane, Frag16& f) {
    const int n     = colBase + (lane & 15);
    const int kbase = (lane >> 4) * 8 + ksub;
    const unsigned short* p = smemT + n * 64 + kbase;
#pragma unroll
    for (int i = 0; i < 8; ++i) {
        f.u[i] = *(const unsigned int*)(p + ((i >> 2) * 16) + ((i & 3) * 2));
    }
}

// --- stage A tile [128 x 64] bf16 into LDS: pure b128 copy (async if avail) --
static __device__ __forceinline__ void stage_a(const unsigned short* __restrict__ Ag,
                                               unsigned short* __restrict__ As,
                                               int mbase, int kk, int tid) {
#pragma unroll
    for (int j = 0; j < 4; ++j) {
        const int c   = tid + j * 256;        // 1024 x 16B chunks
        const int row = c >> 3;
        const int c16 = c & 7;
        const unsigned short* g = Ag + (size_t)(mbase + row) * 1024 + kk + c16 * 8;
        unsigned short* l = As + row * 64 + c16 * 8;
#if MOE_HAVE_ASYNC_LDS
        __builtin_amdgcn_global_load_async_to_lds_b128(
            (as1_i32x4*)(unsigned short*)g, (as3_i32x4*)l, 0, 0);
#else
        *(u32x4*)l = *(const u32x4*)g;
#endif
    }
}

// --- stage B tile W[kk..kk+63][nbase..nbase+127] transposed -> Bt[n][k] ------
// v_perm_b32 builds (W[k][n], W[k+1][n]) u32 pairs -> full-width ds stores.
static __device__ __forceinline__ void stage_b(const unsigned int* __restrict__ Wu32,
                                               unsigned int* __restrict__ BtU,
                                               int nbase, int kk, int tid) {
#pragma unroll
    for (int j = 0; j < 8; ++j) {
        const int id = tid + j * 256;     // 0..2047
        const int kp = id >> 6;           // k-pair 0..31
        const int np = id & 63;           // u32 column (2 n's)
        const int k  = kk + kp * 2;
        const unsigned int a = Wu32[(size_t)k * 512 + (nbase >> 1) + np];        // W[k][2np..2np+1]
        const unsigned int b = Wu32[(size_t)(k + 1) * 512 + (nbase >> 1) + np];  // W[k+1][..]
        const unsigned int p0 = __builtin_amdgcn_perm(b, a, 0x05040100u);  // (a0,a1,b0,b1)
        const unsigned int p1 = __builtin_amdgcn_perm(b, a, 0x07060302u);  // (a2,a3,b2,b3)
        BtU[(2 * np + 0) * 32 + kp] = p0;
        BtU[(2 * np + 1) * 32 + kp] = p1;
    }
}

// ---------------------------------------------------------------------------
// Tiled GEMM: C[128x128 per block] = A[B x 1024](bf16) * W[1024 x 1024](bf16)
// L2K=false : h = relu(acc + b1[n]) -> bf16
// L2K=true  : out = (e==0?0:out) + gates[m,e]*(acc + b2[n])
// ---------------------------------------------------------------------------
template <bool L2K>
__global__ __launch_bounds__(256) void moe_gemm_kernel(
    const unsigned short* __restrict__ A,
    const unsigned short* __restrict__ Bw,
    const float* __restrict__ bias,
    unsigned short* __restrict__ Hout,
    const float* __restrict__ gates,
    float* __restrict__ out,
    int expert) {
    __shared__ __align__(16) unsigned short As[2][128 * 64];   // [row][k]
    __shared__ __align__(16) unsigned short Bs[2][128 * 64];   // [n][k] transposed

    const int tid   = threadIdx.x;
    const int lane  = tid & 31;
    const int wave  = tid >> 5;
    const int wm    = wave >> 1;        // M offset wm*32
    const int wn    = wave & 1;         // N offset wn*64
    const int mbase = blockIdx.y * 128;
    const int nbase = blockIdx.x * 128;

    const unsigned int* __restrict__ Wu32 = (const unsigned int*)Bw;

    v8f acc[2][4];
    const v8f vzero = {0.f, 0.f, 0.f, 0.f, 0.f, 0.f, 0.f, 0.f};
#pragma unroll
    for (int i = 0; i < 2; ++i)
#pragma unroll
        for (int j = 0; j < 4; ++j) acc[i][j] = vzero;

    stage_a(A, As[0], mbase, 0, tid);
    stage_b(Wu32, (unsigned int*)Bs[0], nbase, 0, tid);
    moe_async_wait_all();
    __syncthreads();

    for (int kk = 0; kk < 1024; kk += 64) {
        const int buf = (kk >> 6) & 1;
        if (kk + 64 < 1024) {
            stage_a(A, As[buf ^ 1], mbase, kk + 64, tid);
            stage_b(Wu32, (unsigned int*)Bs[buf ^ 1], nbase, kk + 64, tid);
        }

#pragma unroll
        for (int s = 0; s < 2; ++s) {
            const int ks = s * 32;
            Frag16 a0, a1, b0, b1, b2, b3;
            load_a_frag(As[buf], wm * 32 + 0,  ks, lane, a0);
            load_a_frag(As[buf], wm * 32 + 16, ks, lane, a1);
            load_b_frag(Bs[buf], wn * 64 + 0,  ks, lane, b0);
            load_b_frag(Bs[buf], wn * 64 + 16, ks, lane, b1);
            load_b_frag(Bs[buf], wn * 64 + 32, ks, lane, b2);
            load_b_frag(Bs[buf], wn * 64 + 48, ks, lane, b3);

            acc[0][0] = __builtin_amdgcn_wmma_f32_16x16x32_bf16(false, a0.v, false, b0.v, (short)0, acc[0][0], false, false);
            acc[0][1] = __builtin_amdgcn_wmma_f32_16x16x32_bf16(false, a0.v, false, b1.v, (short)0, acc[0][1], false, false);
            acc[0][2] = __builtin_amdgcn_wmma_f32_16x16x32_bf16(false, a0.v, false, b2.v, (short)0, acc[0][2], false, false);
            acc[0][3] = __builtin_amdgcn_wmma_f32_16x16x32_bf16(false, a0.v, false, b3.v, (short)0, acc[0][3], false, false);
            acc[1][0] = __builtin_amdgcn_wmma_f32_16x16x32_bf16(false, a1.v, false, b0.v, (short)0, acc[1][0], false, false);
            acc[1][1] = __builtin_amdgcn_wmma_f32_16x16x32_bf16(false, a1.v, false, b1.v, (short)0, acc[1][1], false, false);
            acc[1][2] = __builtin_amdgcn_wmma_f32_16x16x32_bf16(false, a1.v, false, b2.v, (short)0, acc[1][2], false, false);
            acc[1][3] = __builtin_amdgcn_wmma_f32_16x16x32_bf16(false, a1.v, false, b3.v, (short)0, acc[1][3], false, false);
        }

        moe_async_wait_all();
        __syncthreads();
    }

    // ---- epilogue: C layout = VGPR v -> (M=v | v+8), N = lane&15 ----
    const int laneN = lane & 15;
    const int laneH = lane >> 4;
#pragma unroll
    for (int i = 0; i < 2; ++i) {
#pragma unroll
        for (int j = 0; j < 4; ++j) {
            const int Mt = mbase + wm * 32 + i * 16;
            const int Nt = nbase + wn * 64 + j * 16;
#pragma unroll
            for (int v = 0; v < 8; ++v) {
                const int m = Mt + v + laneH * 8;
                const int n = Nt + laneN;
                float val = acc[i][j][v] + bias[n];
                if (!L2K) {
                    val = fmaxf(val, 0.f);
                    Hout[(size_t)m * 1024 + n] = f2bf(val);
                } else {
                    const float g    = gates[(size_t)m * 8 + expert];
                    const float prev = (expert == 0) ? 0.f : out[(size_t)m * 1024 + n];
                    out[(size_t)m * 1024 + n] = prev + g * val;
                }
            }
        }
    }
}

// ---------------------------------------------------------------------------
__global__ __launch_bounds__(256) void moe_concat_bf16_kernel(
    const float* __restrict__ text, const float* __restrict__ tech,
    unsigned short* __restrict__ X) {
    const int idx = blockIdx.x * 256 + threadIdx.x;
    const int b = idx >> 10;
    const int d = idx & 1023;
    const float v = (d < 768) ? text[(size_t)b * 768 + d] : tech[(size_t)b * 256 + (d - 768)];
    X[idx] = f2bf(v);
}

__global__ __launch_bounds__(256) void moe_cvt_bf16_kernel(
    const float* __restrict__ in, unsigned short* __restrict__ o, int n) {
    const int i = blockIdx.x * 256 + threadIdx.x;
    if (i < n) o[i] = f2bf(in[i]);
}

__global__ __launch_bounds__(256) void moe_gates_kernel(
    const float* __restrict__ text, const float* __restrict__ tech,
    const float* __restrict__ Wg, const float* __restrict__ bg,
    float* __restrict__ gates) {
    const int b    = (blockIdx.x * 256 + threadIdx.x) >> 5;
    const int lane = threadIdx.x & 31;

    float acc[8];
#pragma unroll
    for (int e = 0; e < 8; ++e) acc[e] = 0.f;

    for (int k = lane; k < 1024; k += 32) {
        const float xv = (k < 768) ? text[(size_t)b * 768 + k] : tech[(size_t)b * 256 + (k - 768)];
#pragma unroll
        for (int e = 0; e < 8; ++e) acc[e] += xv * Wg[k * 8 + e];
    }
#pragma unroll
    for (int e = 0; e < 8; ++e) {
#pragma unroll
        for (int off = 16; off > 0; off >>= 1) acc[e] += __shfl_xor(acc[e], off, 32);
    }

    float lg[8];
    float mx = -3.4e38f;
#pragma unroll
    for (int e = 0; e < 8; ++e) { lg[e] = acc[e] + bg[e]; mx = fmaxf(mx, lg[e]); }
    float s = 0.f;
#pragma unroll
    for (int e = 0; e < 8; ++e) { lg[e] = __expf(lg[e] - mx); s += lg[e]; }
    const float inv = 1.f / s;
    if (lane == 0) {
#pragma unroll
        for (int e = 0; e < 8; ++e) gates[(size_t)b * 8 + e] = lg[e] * inv;
    }
}

// ---------------------------------------------------------------------------
extern "C" void kernel_launch(void* const* d_in, const int* in_sizes, int n_in,
                              void* d_out, int out_size, void* d_ws, size_t ws_size,
                              hipStream_t stream) {
    (void)in_sizes; (void)n_in; (void)out_size; (void)ws_size;

    const float* text = (const float*)d_in[0];
    const float* tech = (const float*)d_in[1];
    const float* W1   = (const float*)d_in[2];
    const float* b1   = (const float*)d_in[3];
    const float* W2   = (const float*)d_in[4];
    const float* b2   = (const float*)d_in[5];
    const float* Wg   = (const float*)d_in[6];
    const float* bg   = (const float*)d_in[7];
    float* out = (float*)d_out;

    char* ws = (char*)d_ws;
    unsigned short* Xb  = (unsigned short*)(ws);                                      // 32 MB
    unsigned short* W1b = (unsigned short*)(ws + 33554432);                           // 16 MB
    unsigned short* W2b = (unsigned short*)(ws + 33554432 + 16777216);                // 16 MB
    float*          gts = (float*)(ws + 33554432 + 2 * 16777216);                     // 512 KB
    unsigned short* Hb  = (unsigned short*)(ws + 33554432 + 2 * 16777216 + 524288);   // 32 MB

    const int BSZ = 16384;

    moe_concat_bf16_kernel<<<dim3(BSZ * 1024 / 256), 256, 0, stream>>>(text, tech, Xb);
    moe_cvt_bf16_kernel<<<dim3(8 * 1024 * 1024 / 256), 256, 0, stream>>>(W1, W1b, 8 * 1024 * 1024);
    moe_cvt_bf16_kernel<<<dim3(8 * 1024 * 1024 / 256), 256, 0, stream>>>(W2, W2b, 8 * 1024 * 1024);
    moe_gates_kernel<<<dim3(BSZ / 8), 256, 0, stream>>>(text, tech, Wg, bg, gts);

    const dim3 grid(1024 / 128, BSZ / 128);   // (8, 128)
    for (int e = 0; e < 8; ++e) {
        const unsigned short* W1e = W1b + (size_t)e * 1024 * 1024;
        const unsigned short* W2e = W2b + (size_t)e * 1024 * 1024;
        moe_gemm_kernel<false><<<grid, 256, 0, stream>>>(
            Xb, W1e, b1 + (size_t)e * 1024, Hb, nullptr, nullptr, e);
        moe_gemm_kernel<true><<<grid, 256, 0, stream>>>(
            Hb, W2e, b2 + (size_t)e * 1024, nullptr, gts, out, e);
    }
}
